// MultiHeadAttention_87935160418612
// MI455X (gfx1250) — matrix operand order repaired and checked by reading, compiled, not run on hardware
//
#include <hip/hip_runtime.h>
#include <hip/hip_bf16.h>

// MHA forward for MI455X (gfx1250, wave32, WMMA + TDM).
// Compute-bound (~137 GFLOP vs ~0.25GB HBM traffic) -> all GEMMs via
// v_wmma_f32_16x16x32_f16 with fp32 accumulation. The reference mask is
// causal (tril), so it is applied analytically instead of streaming 67MB.
// Flash-attention K tiles are DMA'd global->LDS by the Tensor Data Mover
// (tensor_load_to_lds + s_wait_tensorcnt), V tiles go through the manual
// transposed-store path (TDM cannot transpose).

typedef __attribute__((ext_vector_type(16))) _Float16 v16h;
typedef __attribute__((ext_vector_type(8)))  float    v8f;

#define WMMA_F16(A, Bf, C) \
  __builtin_amdgcn_wmma_f32_16x16x32_f16(false, (A), false, (Bf), (short)0, (C), false, false)

#if defined(__gfx1250__) && __has_builtin(__builtin_amdgcn_tensor_load_to_lds) && \
    __has_builtin(__builtin_amdgcn_s_wait_tensorcnt)
#define USE_TDM 1
#else
#define USE_TDM 0
#endif

namespace {
constexpr int kB   = 4;
constexpr int kS   = 2048;
constexpr int kD   = 1024;
constexpr int kH   = 16;
constexpr int kHD  = 64;
constexpr int kM   = kB * kS;        // 8192 GEMM rows
constexpr int QTILE  = 128;          // queries per block (8 waves x 16)
constexpr int KVTILE = 64;           // keys per LDS tile
}

#if USE_TDM
typedef unsigned int u32x4 __attribute__((ext_vector_type(4)));
typedef int          i32x4 __attribute__((ext_vector_type(4)));
typedef int          i32x8 __attribute__((ext_vector_type(8)));

// D# for a 64x64 f16 tile of a [2048 x 64] row-major tensor (CDNA5 ISA):
//  group0: count=1 | lds_addr | global_addr[56:0] | type=2
//  group1: data_size=1(2B); tensor_dim0=64 (bits79:48); tensor_dim1=2048
//          (bits111:80); tile_dim0=64 (bits127:112); tile_dim1=64
//          (bits143:128); tensor_dim0_stride=64 (bits207:160)
__device__ inline void tdm_load_kv_tile(unsigned lds_off, const _Float16* gsrc) {
  const unsigned long long ga = (unsigned long long)(uintptr_t)gsrc;
  u32x4 g0 = { 1u, lds_off, (unsigned)ga,
               (unsigned)((ga >> 32) & 0x1FFFFFFu) | (2u << 30) };
  i32x8 g1 = { 0x00010000, 0x00400000, 0x08000000, 0x00400000, 64, 64, 0, 0 };
  i32x4 z4 = { 0, 0, 0, 0 };
#if __has_include(<hip/amd_detail/amd_gfx1250_TDM.h>)
  i32x8 z8 = { 0, 0, 0, 0, 0, 0, 0, 0 };
  __builtin_amdgcn_tensor_load_to_lds(g0, g1, z4, z4, z8, 0);   // clang-23 form
#else
  __builtin_amdgcn_tensor_load_to_lds(g0, g1, z4, z4, 0);       // ROCm 7.2 form
#endif
}
#endif  // USE_TDM

// ---------------- A/B fragment loaders (contiguous k order, same for A and B) ----------------
// A (16x32 f16, MxK): lane l -> row M=l%16; base=(l/16)*16;
//   element e in [0,16): K=base+e
__device__ inline v16h load_a_f32(const float* __restrict__ p /*row base + k0 + abase*/) {
  v16h a;
#pragma unroll
  for (int i = 0; i < 8; ++i) { a[i] = (_Float16)p[i]; a[i + 8] = (_Float16)p[8 + i]; }
  return a;
}
__device__ inline v16h load_a_f16(const _Float16* __restrict__ p) {
  v16h a;
#pragma unroll
  for (int i = 0; i < 8; ++i) { a[i] = p[i]; a[i + 8] = p[8 + i]; }
  return a;
}
// B (32x16 f16, KxN): lane l -> col N=l%16; element e -> K = 16*(l/16) + e
__device__ inline v16h load_b_f32(const float* __restrict__ p /*col base + k0 + 16*grp*/) {
  v16h b;
#pragma unroll
  for (int i = 0; i < 16; ++i) b[i] = (_Float16)p[i];
  return b;
}
__device__ inline v16h load_b_f16(const _Float16* __restrict__ p) {
  v16h b;
#pragma unroll
  for (int i = 0; i < 16; ++i) b[i] = p[i];
  return b;
}

// ---------------- QKV projection: Y[b*H+h][s][d] = f16(X @ W^T + bias) -----
__global__ void __launch_bounds__(256)
mha_proj_qkv_kernel(const float* __restrict__ X,   // [M, K=1024]
                    const float* __restrict__ W,   // [N=1024, K=1024] (nn.Linear)
                    const float* __restrict__ bias,
                    _Float16* __restrict__ Y) {    // [B*H, S, 64]
  const int lane = threadIdx.x & 31;
  const int wave = threadIdx.x >> 5;
  const int grp  = lane >> 4;
  const int l16  = lane & 15;
  const int m0   = blockIdx.x * 128 + wave * 16;   // 8 waves cover 128 rows
  const int n0   = blockIdx.y * 64;
  const int mrow = m0 + l16;
  const int abase = grp * 16;

  v8f acc[4] = {};
  const float* arow = X + (size_t)mrow * kD + abase;
  for (int k0 = 0; k0 < kD; k0 += 32) {
    v16h a = load_a_f32(arow + k0);
#pragma unroll
    for (int sub = 0; sub < 4; ++sub) {
      const int n = n0 + sub * 16 + l16;
      v16h bf = load_b_f32(W + (size_t)n * kD + k0 + grp * 16);
      acc[sub] = WMMA_F16(a, bf, acc[sub]);
    }
  }
  // C layout: row = r + 8*grp, col = l16 (+16*sub). Scatter to [B*H, S, 64].
#pragma unroll
  for (int sub = 0; sub < 4; ++sub) {
    const int n = n0 + sub * 16 + l16;
    const int h = n >> 6, d = n & 63;
    const float bv = bias[n];
#pragma unroll
    for (int r = 0; r < 8; ++r) {
      const int m = m0 + r + grp * 8;
      const int bb = m / kS, s = m % kS;
      Y[(((size_t)bb * kH + h) * kS + s) * kHD + d] = (_Float16)(acc[sub][r] + bv);
    }
  }
}

// ---------------- Flash attention (causal, online softmax) ----------------
__global__ void __launch_bounds__(256)
mha_flash_attn_kernel(const _Float16* __restrict__ Qw,  // [B*H, S, 64]
                      const _Float16* __restrict__ Kw,
                      const _Float16* __restrict__ Vw,
                      _Float16* __restrict__ Ow) {      // [B, S, D]
  __shared__ _Float16 Kt[KVTILE][kHD];       // [key][hd]   (B frag for QK^T)
  __shared__ _Float16 Vt[kHD][KVTILE + 8];   // [hd][key]   (B frag for PV)
  __shared__ _Float16 Pt[8][16][KVTILE];     // per-wave P, C->A layout bridge

  const int lane = threadIdx.x & 31;
  const int wave = threadIdx.x >> 5;
  const int grp  = lane >> 4;
  const int l16  = lane & 15;
  const int bh   = blockIdx.y;               // b*H + h
  const int q0   = blockIdx.x * QTILE;
  const int abase = grp * 16;

  // Q fragments for this wave's 16 rows (A layout), hd = 64 -> 2 chunks of 32
  const int qrow = q0 + wave * 16 + l16;
  const _Float16* qp = Qw + ((size_t)bh * kS + qrow) * kHD;
  v16h qf[2];
#pragma unroll
  for (int c = 0; c < 2; ++c) qf[c] = load_a_f16(qp + c * 32 + abase);

  float mrow[8], lrow[8];
  v8f accO[4] = {};
#pragma unroll
  for (int r = 0; r < 8; ++r) { mrow[r] = -1.0e30f; lrow[r] = 0.0f; }

  const float scale = 0.125f;                // 1/sqrt(64)
  const size_t kvbase = (size_t)bh * kS * kHD;
  const int kv_end = (q0 + QTILE < kS) ? (q0 + QTILE) : kS;  // causal bound

  for (int kv0 = 0; kv0 < kv_end; kv0 += KVTILE) {
#if USE_TDM
    // K tile: one Tensor Data Mover DMA per tile, issued by wave 0 right
    // after the barrier that retired all readers of Kt.
    if (wave == 0) {
      tdm_load_kv_tile((unsigned)(uintptr_t)&Kt[0][0],
                       Kw + kvbase + (size_t)kv0 * kHD);
    }
#endif
    // V tile: cooperative coalesced load + transpose into [hd][key].
    {
      const int t = threadIdx.x;
#pragma unroll
      for (int i = 0; i < 16; ++i) {
        const int idx = t * 16 + i;
        const int key = idx >> 6, hd = idx & 63;
        const size_t g = kvbase + (size_t)(kv0 + key) * kHD + hd;
#if !USE_TDM
        Kt[key][hd] = Kw[g];
#endif
        Vt[hd][key] = Vw[g];
      }
      if (kv0 + KVTILE < kv_end) {  // prefetch next tile -> global_prefetch_b8
        const size_t gn = kvbase + (size_t)(kv0 + KVTILE) * kHD + (size_t)t * 16;
        __builtin_prefetch(Kw + gn, 0, 1);
        __builtin_prefetch(Vw + gn, 0, 1);
      }
    }
#if USE_TDM
    if (wave == 0) __builtin_amdgcn_s_wait_tensorcnt(0);
#endif
    __syncthreads();

    // S = Q K^T  (16 q x 64 keys per wave)
    v8f accS[4] = {};
#pragma unroll
    for (int c = 0; c < 2; ++c)
#pragma unroll
      for (int sub = 0; sub < 4; ++sub) {
        v16h bf = load_b_f16(&Kt[sub * 16 + l16][c * 32 + grp * 16]);
        accS[sub] = WMMA_F16(qf[c], bf, accS[sub]);
      }

    // scale + causal mask + online softmax (rows live in 16-lane groups)
    float pcur[4][8], rm[8];
#pragma unroll
    for (int r = 0; r < 8; ++r) rm[r] = -1.0e30f;
#pragma unroll
    for (int sub = 0; sub < 4; ++sub) {
      const int col = kv0 + sub * 16 + l16;
#pragma unroll
      for (int r = 0; r < 8; ++r) {
        const int row = q0 + wave * 16 + r + grp * 8;
        float sv = accS[sub][r] * scale;
        if (col > row) sv = -1.0e30f;
        pcur[sub][r] = sv;
        rm[r] = fmaxf(rm[r], sv);
      }
    }
#pragma unroll
    for (int r = 0; r < 8; ++r) {
#pragma unroll
      for (int off = 1; off < 16; off <<= 1)
        rm[r] = fmaxf(rm[r], __shfl_xor(rm[r], off, 32));
      const float mnew = fmaxf(mrow[r], rm[r]);
      const float alpha = __expf(mrow[r] - mnew);
      mrow[r] = mnew;
      float rs = 0.0f;
#pragma unroll
      for (int sub = 0; sub < 4; ++sub) {
        const float p = __expf(pcur[sub][r] - mnew);
        pcur[sub][r] = p;
        rs += p;
      }
#pragma unroll
      for (int off = 1; off < 16; off <<= 1) rs += __shfl_xor(rs, off, 32);
      lrow[r] = lrow[r] * alpha + rs;
#pragma unroll
      for (int sub = 0; sub < 4; ++sub) accO[sub][r] *= alpha;
    }

    // P: C layout -> LDS -> A layout
#pragma unroll
    for (int sub = 0; sub < 4; ++sub)
#pragma unroll
      for (int r = 0; r < 8; ++r)
        Pt[wave][r + grp * 8][sub * 16 + l16] = (_Float16)pcur[sub][r];
    __syncthreads();

    // O += P @ V
#pragma unroll
    for (int c = 0; c < 2; ++c) {
      v16h af = load_a_f16(&Pt[wave][l16][c * 32 + abase]);
#pragma unroll
      for (int sub = 0; sub < 4; ++sub) {
        v16h bf = load_b_f16(&Vt[sub * 16 + l16][c * 32 + grp * 16]);
        accO[sub] = WMMA_F16(af, bf, accO[sub]);
      }
    }
    __syncthreads();
  }

  // epilogue: normalize and store f16 to [B, S, D] (D = h*64 + d)
  const int bb = bh / kH, h = bh % kH;
#pragma unroll
  for (int r = 0; r < 8; ++r) {
    const float inv = 1.0f / lrow[r];
    const int row = q0 + wave * 16 + r + grp * 8;
#pragma unroll
    for (int sub = 0; sub < 4; ++sub) {
      const int d = sub * 16 + l16;
      Ow[((size_t)bb * kS + row) * kD + h * kHD + d] = (_Float16)(accO[sub][r] * inv);
    }
  }
}

// ---------------- Output projection: out = O @ Wo^T + bo (f32 out) --------
__global__ void __launch_bounds__(256)
mha_out_proj_kernel(const _Float16* __restrict__ O,  // [M, 1024] f16
                    const float* __restrict__ W,     // [1024, 1024]
                    const float* __restrict__ bias,
                    float* __restrict__ Y) {         // [M, 1024] f32
  const int lane = threadIdx.x & 31;
  const int wave = threadIdx.x >> 5;
  const int grp  = lane >> 4;
  const int l16  = lane & 15;
  const int m0   = blockIdx.x * 128 + wave * 16;
  const int n0   = blockIdx.y * 64;
  const int mrow = m0 + l16;
  const int abase = grp * 16;

  v8f acc[4] = {};
  const _Float16* arow = O + (size_t)mrow * kD + abase;
  for (int k0 = 0; k0 < kD; k0 += 32) {
    v16h a = load_a_f16(arow + k0);
#pragma unroll
    for (int sub = 0; sub < 4; ++sub) {
      const int n = n0 + sub * 16 + l16;
      v16h bf = load_b_f32(W + (size_t)n * kD + k0 + grp * 16);
      acc[sub] = WMMA_F16(a, bf, acc[sub]);
    }
  }
#pragma unroll
  for (int sub = 0; sub < 4; ++sub) {
    const int n = n0 + sub * 16 + l16;
    const float bv = bias[n];
#pragma unroll
    for (int r = 0; r < 8; ++r) {
      const int m = m0 + r + grp * 8;
      Y[(size_t)m * kD + n] = acc[sub][r] + bv;
    }
  }
}

extern "C" void kernel_launch(void* const* d_in, const int* in_sizes, int n_in,
                              void* d_out, int out_size, void* d_ws, size_t ws_size,
                              hipStream_t stream) {
  (void)in_sizes; (void)n_in; (void)out_size; (void)ws_size;
  const float* q  = (const float*)d_in[0];
  const float* k  = (const float*)d_in[1];
  const float* v  = (const float*)d_in[2];
  // d_in[3] = causal mask (tril) — applied analytically in the kernel.
  const float* Wq = (const float*)d_in[4];  const float* bq = (const float*)d_in[5];
  const float* Wk = (const float*)d_in[6];  const float* bk = (const float*)d_in[7];
  const float* Wv = (const float*)d_in[8];  const float* bv = (const float*)d_in[9];
  const float* Wo = (const float*)d_in[10]; const float* bo = (const float*)d_in[11];
  float* out = (float*)d_out;

  _Float16* ws = (_Float16*)d_ws;
  const size_t NQ = (size_t)kB * kH * kS * kHD;  // 8.4M halves per tensor
  _Float16* q_ws = ws;
  _Float16* k_ws = ws + NQ;
  _Float16* v_ws = ws + 2 * NQ;
  _Float16* o_ws = ws + 3 * NQ;

  dim3 gemm_grid(kM / 128, kD / 64);   // 64 x 16
  dim3 attn_grid(kS / QTILE, kB * kH); // 16 x 64

  mha_proj_qkv_kernel<<<gemm_grid, 256, 0, stream>>>(q, Wq, bq, q_ws);
  mha_proj_qkv_kernel<<<gemm_grid, 256, 0, stream>>>(k, Wk, bk, k_ws);
  mha_proj_qkv_kernel<<<gemm_grid, 256, 0, stream>>>(v, Wv, bv, v_ws);
  mha_flash_attn_kernel<<<attn_grid, 256, 0, stream>>>(q_ws, k_ws, v_ws, o_ws);
  mha_out_proj_kernel<<<gemm_grid, 256, 0, stream>>>(o_ws, Wo, bo, out);
}
